// HierCVQLayer_65824668779019
// MI455X (gfx1250) — compile-verified, matrix operand order checked by reading, not run on hardware
//
#include <hip/hip_runtime.h>
#include <math.h>
#include <stdint.h>

// ---------------- CDNA5 WMMA types ----------------
typedef __attribute__((ext_vector_type(16))) __bf16 v16bf;
typedef __attribute__((ext_vector_type(8)))  float  v8f;

#define NTOK   32768
#define EDIM   2048
#define VDIM   256
#define KCODE  256
#define HID    1024

__device__ __forceinline__ uint32_t hashu(uint32_t x) {
  x ^= x >> 17; x *= 0xED5AD4BBu; x ^= x >> 11; x *= 0xAC4C1B51u;
  x ^= x >> 15; x *= 0x31848BABu; x ^= x >> 14; return x;
}

__device__ __forceinline__ v8f wmma_bf16(v16bf a, v16bf b, v8f c) {
  return __builtin_amdgcn_wmma_f32_16x16x32_bf16(false, a, false, b, (short)0, c, false, false);
}

// Stage one 16x32 A-chunk (rows m0..m0+15, k = kc*32..+31) from f32 global into
// LDS as bf16 in exact WMMA A-fragment order: buf[lane*16 + e] =
// A[lane&15][kc*32 + e + 8*(e>>3) + 8*(lane>>4)].  512 elems, 2 per thread.
__device__ __forceinline__ void stage_a(const float* __restrict__ X, int ldk,
                                        int m0, int kc, __bf16* buf, int t) {
#pragma unroll
  for (int s = 0; s < 2; ++s) {
    int j = t + 256 * s;
    int lane = j >> 4, e = j & 15;
    int row = lane & 15;
    int k = kc * 32 + e + ((e >> 3) << 3) + ((lane >> 4) << 3);
    buf[j] = (__bf16)X[(size_t)(m0 + row) * ldk + k];
  }
}

// ---------------- generic WMMA GEMM: Y = X @ W (+bias) --------------------
// X: [M,K] f32 (M = 16*gridDim.y), Wp: packed bf16 fragments, Y: [M,N] f32.
// Block: 256 threads = 8 waves; block tile 16 x 256 (wave w -> cols 32w..32w+31).
__global__ __launch_bounds__(256) void wmma_gemm(
    const float* __restrict__ X, const __bf16* __restrict__ Wp,
    const float* __restrict__ bias, float* __restrict__ Y, int K, int N) {
  __shared__ __attribute__((aligned(32))) __bf16 hAp[2][512];
  const int t    = threadIdx.x;
  const int lane = t & 31;
  const int w    = t >> 5;
  const int m0   = blockIdx.y * 16;
  const int n0   = blockIdx.x * 256;
  const int NT   = N >> 4;
  const int KC   = K >> 5;
  const v16bf* pwv = (const v16bf*)Wp;

  v8f acc0 = {}; v8f acc1 = {};
  stage_a(X, K, m0, 0, hAp[0], t);
  for (int kc = 0; kc < KC; ++kc) {
    __syncthreads();
    v16bf a = ((const v16bf*)hAp[kc & 1])[lane];
    if (kc + 1 < KC) stage_a(X, K, m0, kc + 1, hAp[(kc + 1) & 1], t);
    int nt = (n0 >> 4) + 2 * w;
    v16bf b0 = pwv[(size_t)(kc * NT + nt) * 32 + lane];
    v16bf b1 = pwv[(size_t)(kc * NT + nt + 1) * 32 + lane];
    acc0 = wmma_bf16(a, b0, acc0);
    acc1 = wmma_bf16(a, b1, acc1);
  }
  int cb = n0 + w * 32 + (lane & 15);
  float b0v = bias ? bias[cb] : 0.0f;
  float b1v = bias ? bias[cb + 16] : 0.0f;
#pragma unroll
  for (int r = 0; r < 8; ++r) {
    int row = m0 + r + (lane >> 4) * 8;
    Y[(size_t)row * N + cb]      = acc0[r] + b0v;
    Y[(size_t)row * N + cb + 16] = acc1[r] + b1v;
  }
}

// ---------------- codebook first layer (K=8 bits, trivial) ----------------
__global__ __launch_bounds__(256) void gemm_bits(const float* __restrict__ w0,
                                                 const float* __restrict__ b0,
                                                 float* __restrict__ Y) {
  int idx = blockIdx.x * 256 + threadIdx.x;        // 256*1024 elements
  int r = idx >> 10, c = idx & 1023;
  float s = b0[c];
#pragma unroll
  for (int j = 0; j < 8; ++j)
    if ((r >> (7 - j)) & 1) s += w0[j * HID + c];
  Y[idx] = s;
}

// BatchNorm (training stats over 256 rows) + ReLU, one column per block
__global__ __launch_bounds__(256) void bn_relu(float* __restrict__ X,
                                               const float* __restrict__ g,
                                               const float* __restrict__ be,
                                               int C) {
  __shared__ float sm[256], sv[256];
  int c = blockIdx.x, r = threadIdx.x;
  float x = X[(size_t)r * C + c];
  sm[r] = x; sv[r] = x * x;
  __syncthreads();
  for (int o = 128; o > 0; o >>= 1) {
    if (r < o) { sm[r] += sm[r + o]; sv[r] += sv[r + o]; }
    __syncthreads();
  }
  float m   = sm[0] * (1.0f / 256.0f);
  float var = sv[0] * (1.0f / 256.0f) - m * m;
  float y = g[c] * (x - m) * rsqrtf(var + 1e-5f) + be[c];
  X[(size_t)r * C + c] = fmaxf(y, 0.0f);
}

__global__ __launch_bounds__(256) void row_normalize(const float* __restrict__ src,
                                                     float* __restrict__ dst) {
  __shared__ float s[256];
  int row = blockIdx.x, t = threadIdx.x;
  float x = src[(size_t)row * 256 + t];
  s[t] = x * x;
  __syncthreads();
  for (int o = 128; o > 0; o >>= 1) {
    if (t < o) s[t] += s[t + o];
    __syncthreads();
  }
  float inv = 1.0f / (sqrtf(s[0]) + 1e-6f);
  dst[(size_t)row * 256 + t] = x * inv;
}

// ---------------- pack B matrix into WMMA bf16 fragment order ----------------
// packed[((kc*NT + nt)*32 + lane)*16 + e] = B[kc*32 + e + 8*(e>>3) + 8*(lane>>4)][nt*16 + (lane&15)]
// transposed==0: B[k][n] = src[k*N + n];  transposed==1: B[k][n] = src[n*K + k]
__global__ __launch_bounds__(256) void pack_b(const float* __restrict__ src,
                                              __bf16* __restrict__ dst,
                                              int K, int N, int transposed) {
  int idx = blockIdx.x * 256 + threadIdx.x;
  if (idx >= K * N) return;
  int e    = idx & 15;
  int lane = (idx >> 4) & 31;
  int fr   = idx >> 9;
  int NT   = N >> 4;
  int kc = fr / NT, nt = fr % NT;
  int k = kc * 32 + e + ((e >> 3) << 3) + ((lane >> 4) << 3);
  int n = nt * 16 + (lane & 15);
  float v = transposed ? src[(size_t)n * K + k] : src[(size_t)k * N + n];
  dst[idx] = (__bf16)v;
}

__global__ void zero1(float* p) { if (threadIdx.x == 0) *p = 0.0f; }

// ---------------- fused: proj GEMM -> normalize -> sim GEMM -> softmax ->
//                  sample -> probs/quantized/code/loss (16 tokens / block) ----
__global__ __launch_bounds__(256) void fused_vq(
    const float* __restrict__ h_in,        // [N,E]
    const float* __restrict__ proj_b,      // [V]
    const __bf16* __restrict__ pw,         // packed proj_w fragments
    const __bf16* __restrict__ pe,         // packed embed^T fragments
    const float* __restrict__ embed,       // [K,V] normalized
    const float* __restrict__ temp,        // [1]
    float* __restrict__ out_code,          // [N]
    float* __restrict__ out_quant,         // [N,V]
    float* __restrict__ out_probs,         // [N,K]
    float* __restrict__ out_loss,          // [1]
    int*   __restrict__ ws_code,           // [N]
    float* __restrict__ ws_sel)            // [N]
{
  __shared__ __attribute__((aligned(32))) __bf16 hAp[2][512];
  __shared__ float htile[16 * 260];
  __shared__ float ltile[16 * 256];
  __shared__ float red[16 * 16];
  __shared__ float rowval[16];
  __shared__ int   codeSh[16];
  __shared__ float selSh[16];

  const int t    = threadIdx.x;
  const int lane = t & 31;
  const int w    = t >> 5;          // wave 0..7 -> col tiles 2w, 2w+1
  const int half = lane >> 4;
  const int m0   = blockIdx.x * 16;

  const v16bf* pwv = (const v16bf*)pw;
  const v16bf* pev = (const v16bf*)pe;

  // ---- GEMM1: h_pre = h_in[m0:m0+16,:] @ proj_w  (bf16 WMMA, K=2048) ----
  v8f acc0 = {}; v8f acc1 = {};
  stage_a(h_in, EDIM, m0, 0, hAp[0], t);
  for (int kc = 0; kc < EDIM / 32; ++kc) {
    __syncthreads();
    v16bf a = ((const v16bf*)hAp[kc & 1])[lane];
    if (kc + 1 < EDIM / 32) stage_a(h_in, EDIM, m0, kc + 1, hAp[(kc + 1) & 1], t);
    v16bf b0 = pwv[(kc * 16 + 2 * w) * 32 + lane];
    v16bf b1 = pwv[(kc * 16 + 2 * w + 1) * 32 + lane];
    acc0 = wmma_bf16(a, b0, acc0);
    acc1 = wmma_bf16(a, b1, acc1);
  }
  {
    int c0 = w * 32 + (lane & 15);
#pragma unroll
    for (int r = 0; r < 8; ++r) {
      int row = r + half * 8;
      htile[row * 260 + c0]      = acc0[r] + proj_b[c0];
      htile[row * 260 + c0 + 16] = acc1[r] + proj_b[c0 + 16];
    }
  }
  __syncthreads();

  // ---- sphere-normalize the 16 rows in LDS ----
  {
    int row = t >> 4, sub = t & 15;
    float s = 0.0f;
    for (int i = sub; i < VDIM; i += 16) { float x = htile[row * 260 + i]; s += x * x; }
    red[row * 16 + sub] = s;
    __syncthreads();
    if (sub == 0) {
      float tot = 0.0f;
#pragma unroll
      for (int i = 0; i < 16; ++i) tot += red[row * 16 + i];
      rowval[row] = 1.0f / (sqrtf(tot) + 1e-6f);
    }
    __syncthreads();
    float inv = rowval[row];
    for (int i = sub; i < VDIM; i += 16) htile[row * 260 + i] *= inv;
  }
  __syncthreads();

  // ---- GEMM2: logits = 2*alpha * (h @ embed^T)  (bf16 WMMA, K=256) ----
  float alpha = 1.0f / (temp[0] * temp[0]);
  v8f s0 = {}; v8f s1 = {};
#pragma unroll
  for (int kc = 0; kc < VDIM / 32; ++kc) {
    v16bf a;
#pragma unroll
    for (int e = 0; e < 16; ++e) {
      int k = kc * 32 + e + ((e >> 3) << 3) + (half << 3);
      a[e] = (__bf16)htile[(lane & 15) * 260 + k];
    }
    v16bf b0 = pev[(kc * 16 + 2 * w) * 32 + lane];
    v16bf b1 = pev[(kc * 16 + 2 * w + 1) * 32 + lane];
    s0 = wmma_bf16(a, b0, s0);
    s1 = wmma_bf16(a, b1, s1);
  }
  {
    float sc = 2.0f * alpha;
    int c0 = w * 32 + (lane & 15);
#pragma unroll
    for (int r = 0; r < 8; ++r) {
      int row = r + half * 8;
      ltile[row * 256 + c0]      = sc * s0[r];
      ltile[row * 256 + c0 + 16] = sc * s1[r];
    }
  }
  __syncthreads();

  // ---- softmax over 256 codes per row (in place -> probs) ----
  {
    int row = t >> 4, sub = t & 15;
    float mx = -1e30f;
    for (int i = sub; i < KCODE; i += 16) mx = fmaxf(mx, ltile[row * 256 + i]);
    red[row * 16 + sub] = mx;
    __syncthreads();
    if (sub == 0) {
      float m = red[row * 16];
#pragma unroll
      for (int i = 1; i < 16; ++i) m = fmaxf(m, red[row * 16 + i]);
      rowval[row] = m;
    }
    __syncthreads();
    float m = rowval[row];
    float s = 0.0f;
    for (int i = sub; i < KCODE; i += 16) {
      float e = __expf(ltile[row * 256 + i] - m);
      ltile[row * 256 + i] = e;
      s += e;
    }
    red[row * 16 + sub] = s;
    __syncthreads();
    if (sub == 0) {
      float tot = 0.0f;
#pragma unroll
      for (int i = 0; i < 16; ++i) tot += red[row * 16 + i];
      rowval[row] = 1.0f / tot;
    }
    __syncthreads();
    float inv = rowval[row];
    for (int i = sub; i < KCODE; i += 16) ltile[row * 256 + i] *= inv;
  }
  __syncthreads();

  // ---- categorical sample (inverse CDF) + select mask, one thread per row ----
  if (t < 16) {
    int row = t;
    int grow = m0 + row;
    uint32_t hc = hashu(0x9E3779B9u ^ (uint32_t)grow);
    float u = (hc >> 8) * (1.0f / 16777216.0f);
    float cum = 0.0f; int code = KCODE - 1;
    for (int j = 0; j < KCODE; ++j) {
      cum += ltile[row * 256 + j];
      if (u < cum) { code = j; break; }
    }
    uint32_t hp = hashu(0x85EBCA6Bu ^ (uint32_t)grow);
    float p = (hp >> 8) * (1.0f / 16777216.0f);
    float sel = (p > 0.0f && p < 1.0f) ? 1.0f : 0.0f;
    codeSh[row] = code;
    selSh[row]  = sel;
    ws_code[grow] = code;
    ws_sel[grow]  = sel;
    out_code[grow] = (float)code * sel;
  }
  __syncthreads();

  // ---- probs, quantized = sel*embed[code], vq_loss contribution ----
  {
    int row = t >> 4, sub = t & 15;
    int grow = m0 + row;
    int code = codeSh[row];
    float sel = selSh[row];
    float part = 0.0f;
    for (int i = sub; i < VDIM; i += 16) {
      out_probs[(size_t)grow * KCODE + i] = ltile[row * 256 + i] * sel;
      float ev = embed[(size_t)code * VDIM + i];
      out_quant[(size_t)grow * VDIM + i] = ev * sel;
      float d = ev - htile[row * 260 + i];
      part += d * d;
    }
    red[row * 16 + sub] = part;
    __syncthreads();
    if (sub == 0) {
      float tot = 0.0f;
#pragma unroll
      for (int i = 0; i < 16; ++i) tot += red[row * 16 + i];
      atomicAdd(out_loss, sqrtf(tot) * (1.0f / (float)NTOK));
    }
  }
}

// ---------------- quantized_inv = sel*embed_pinv[code] + pinv_b (pure BW) ----
__global__ __launch_bounds__(256) void gather_pinv(const float* __restrict__ ep,   // [K,E], no bias
                                                   const float* __restrict__ pinv_b,
                                                   const int* __restrict__ code,
                                                   const float* __restrict__ sel,
                                                   float* __restrict__ out) {
  size_t idx = (size_t)blockIdx.x * 256 + threadIdx.x;   // float4 granule
  int row = (int)(idx >> 9);                             // E/4 = 512 float4 per row
  int c4  = (int)(idx & 511);
  const float4* ep4 = (const float4*)ep;
  const float4* b4  = (const float4*)pinv_b;
  float4 v  = ep4[(size_t)code[row] * 512 + c4];
  float4 bb = b4[c4];
  float s = sel[row];
  float4 o;
  o.x = s * v.x + bb.x; o.y = s * v.y + bb.y;
  o.z = s * v.z + bb.z; o.w = s * v.w + bb.w;
  ((float4*)out)[idx] = o;
}

extern "C" void kernel_launch(void* const* d_in, const int* in_sizes, int n_in,
                              void* d_out, int out_size, void* d_ws, size_t ws_size,
                              hipStream_t stream) {
  const float* h_in   = (const float*)d_in[0];
  const float* temp   = (const float*)d_in[1];
  const float* proj_w = (const float*)d_in[2];
  const float* proj_b = (const float*)d_in[3];
  const float* pinv_w = (const float*)d_in[4];
  const float* pinv_b = (const float*)d_in[5];
  const float* w0  = (const float*)d_in[6];
  const float* b0  = (const float*)d_in[7];
  const float* g0  = (const float*)d_in[8];
  const float* be0 = (const float*)d_in[9];
  const float* wm  = (const float*)d_in[10];
  const float* bm  = (const float*)d_in[11];
  const float* gm  = (const float*)d_in[12];
  const float* bem = (const float*)d_in[13];
  const float* wL  = (const float*)d_in[14];
  const float* bL  = (const float*)d_in[15];
  float* out = (float*)d_out;

  // d_out layout: quantized_inv[N*E] | vq_code[N] | quantized[N*V] | probs[N*K] | vq_loss[1]
  const size_t o1 = (size_t)NTOK * EDIM;
  const size_t o2 = o1 + NTOK;
  const size_t o3 = o2 + (size_t)NTOK * VDIM;
  const size_t o4 = o3 + (size_t)NTOK * KCODE;

  // workspace carve-up
  char* ws = (char*)d_ws;
  float* X0        = (float*)ws; ws += (size_t)256 * HID * 4;
  float* X1        = (float*)ws; ws += (size_t)256 * HID * 4;
  float* embedraw  = (float*)ws; ws += (size_t)KCODE * VDIM * 4;
  float* embed     = (float*)ws; ws += (size_t)KCODE * VDIM * 4;
  float* embedpinv = (float*)ws; ws += (size_t)KCODE * EDIM * 4;
  __bf16* packedS  = (__bf16*)ws; ws += (size_t)HID * HID * 2;   // reusable pack scratch
  __bf16* packed_pw = (__bf16*)ws; ws += (size_t)EDIM * VDIM * 2;
  __bf16* packed_pe = (__bf16*)ws; ws += (size_t)VDIM * KCODE * 2;
  int*   ws_code = (int*)ws;   ws += (size_t)NTOK * 4;
  float* ws_sel  = (float*)ws; ws += (size_t)NTOK * 4;

  // ---- codebook MLP: first layer from binary codes ----
  gemm_bits<<<(256 * HID) / 256, 256, 0, stream>>>(w0, b0, X0);
  bn_relu<<<HID, 256, 0, stream>>>(X0, g0, be0, HID);

  // ---- 4 middle layers: pack weights -> WMMA GEMM -> BN+ReLU ----
  float* cur = X0; float* nxt = X1;
  for (int i = 0; i < 4; ++i) {
    pack_b<<<(HID * HID) / 256, 256, 0, stream>>>(wm + (size_t)i * HID * HID, packedS,
                                                  HID, HID, 0);
    wmma_gemm<<<dim3(HID / 256, 256 / 16), 256, 0, stream>>>(cur, packedS,
                                                             bm + (size_t)i * HID, nxt,
                                                             HID, HID);
    bn_relu<<<HID, 256, 0, stream>>>(nxt, gm + (size_t)i * HID, bem + (size_t)i * HID, HID);
    float* tmp = cur; cur = nxt; nxt = tmp;
  }

  // ---- final linear -> normalize -> embed ----
  pack_b<<<(HID * VDIM) / 256, 256, 0, stream>>>(wL, packedS, HID, VDIM, 0);
  wmma_gemm<<<dim3(VDIM / 256, 256 / 16), 256, 0, stream>>>(cur, packedS, bL, embedraw,
                                                            HID, VDIM);
  row_normalize<<<KCODE, 256, 0, stream>>>(embedraw, embed);

  // ---- embed_pinv = embed @ pinv_w (bias added in gather) ----
  pack_b<<<(VDIM * EDIM) / 256, 256, 0, stream>>>(pinv_w, packedS, VDIM, EDIM, 0);
  wmma_gemm<<<dim3(EDIM / 256, 256 / 16), 256, 0, stream>>>(embed, packedS, nullptr,
                                                            embedpinv, VDIM, EDIM);

  // ---- pack B matrices for the fused pipeline ----
  pack_b<<<(EDIM * VDIM) / 256, 256, 0, stream>>>(proj_w, packed_pw, EDIM, VDIM, 0);
  pack_b<<<(VDIM * KCODE) / 256, 256, 0, stream>>>(embed, packed_pe, VDIM, KCODE, 1);

  zero1<<<1, 32, 0, stream>>>(out + o4);

  // ---- fused main pipeline: 2048 blocks x 16 tokens ----
  fused_vq<<<NTOK / 16, 256, 0, stream>>>(h_in, proj_b, packed_pw, packed_pe, embed, temp,
                                          out + o1, out + o2, out + o3, out + o4,
                                          ws_code, ws_sel);

  // ---- quantized_inv via table gather (memory-bound) ----
  gather_pinv<<<(int)(((size_t)NTOK * EDIM / 4) / 256), 256, 0, stream>>>(
      embedpinv, pinv_b, ws_code, ws_sel, out);

  (void)in_sizes; (void)n_in; (void)out_size; (void)ws_size;
}
